// PointTransformerV3_55868934586539
// MI455X (gfx1250) — matrix operand order repaired and checked by reading, not compiled, not run on hardware
//
#include <hip/hip_runtime.h>

#define DEVI static __device__ __forceinline__

typedef _Float16 v16h __attribute__((ext_vector_type(16)));
typedef _Float16 v8h  __attribute__((ext_vector_type(8)));
typedef _Float16 v4h  __attribute__((ext_vector_type(4)));
typedef float    v8f  __attribute__((ext_vector_type(8)));
typedef int      v4i  __attribute__((ext_vector_type(4)));

static constexpr int  kN   = 65536;
static constexpr int  kC   = 256;
static constexpr int  kH   = 8;
static constexpr int  kPS  = 128;
static constexpr int  kHID = 1024;
static constexpr int  kP   = kN / kPS;     // 512 patches

struct H16 { v8h lo, hi; };

DEVI v16h mk16(v8h lo, v8h hi) { H16 h{lo, hi}; return __builtin_bit_cast(v16h, h); }

DEVI v8f wmma_f16(v16h a, v16h b, v8f c) {
  return __builtin_amdgcn_wmma_f32_16x16x32_f16(false, a, false, b, (short)0, c,
                                                false, false);
}

// A fragment (16x32 f16), rows row-major with leading dim ld.
// Lane 0-15: M=lane, K = {0..15} in v0..7.
// Lane 16-31: M=lane-16, K = {16..31}.
DEVI v16h load_fragA(const _Float16* base, int ld, int mBase, int kBase, int lane) {
  int half = lane >> 4, mr = lane & 15;
  const _Float16* p = base + (size_t)(mBase + mr) * ld + kBase + half * 16;
  return mk16(*(const v8h*)p, *(const v8h*)(p + 8));
}

// gfx1250 async global->LDS copy (ASYNCcnt-tracked), guarded by probe.
#if defined(__has_builtin)
#if __has_builtin(__builtin_amdgcn_global_load_async_to_lds_b128)
#define HAVE_ASYNC_LDS 1
#endif
#endif

#ifdef HAVE_ASYNC_LDS
typedef __attribute__((address_space(1))) v4i* gv4i_p;   // global (AS1)
typedef __attribute__((address_space(3))) v4i* lv4i_p;   // LDS (AS3)
DEVI void async_copy16B(const _Float16* gsrc, _Float16* ldst) {
  __builtin_amdgcn_global_load_async_to_lds_b128((gv4i_p)(void*)(const void*)gsrc,
                                                 (lv4i_p)(void*)ldst, 0, 0);
}
DEVI void async_wait0() { asm volatile("s_wait_asynccnt 0" ::: "memory"); }
#endif

// ---------------------------------------------------------------------------
// Weight pre-pack into B-fragment order.
// B 32x16: lane 0-15 -> N=lane, K pairs {2j,2j+1} in vj ; lane 16-31 -> K+16.
// Packed offset: ((kt*NT + nt)*32 + lane)*16 + 2j halves.
// ---------------------------------------------------------------------------
__global__ __launch_bounds__(256) void pack_b_kernel(const float* __restrict__ W,
                                                     _Float16* __restrict__ out,
                                                     int K, int Nout) {
  int NT = Nout >> 4;
  size_t tid = (size_t)blockIdx.x * 256 + threadIdx.x;   // K*Nout/2 threads
  int j = tid & 7;
  int lane = (tid >> 3) & 31;
  size_t tile = tid >> 8;                                // kt*NT + nt
  int nt = (int)(tile % NT);
  int kt = (int)(tile / NT);
  int k0 = kt * 32 + (lane >> 4) * 16 + 2 * j;
  int n  = nt * 16 + (lane & 15);
  size_t off = (tile * 32 + lane) * 16 + 2 * j;
  out[off]     = (_Float16)W[(size_t)k0 * Nout + n];
  out[off + 1] = (_Float16)W[(size_t)(k0 + 1) * Nout + n];
}

// ---------------------------------------------------------------------------
// LayerNorm over C=256, one wave per row, f16 output.
// ---------------------------------------------------------------------------
__global__ __launch_bounds__(256) void ln_kernel(const float* __restrict__ x,
                                                 const float* __restrict__ g,
                                                 const float* __restrict__ b,
                                                 _Float16* __restrict__ out) {
  int wave = threadIdx.x >> 5, lane = threadIdx.x & 31;
  int row = blockIdx.x * 8 + wave;
  const float4* xp = (const float4*)(x + (size_t)row * kC);
  float4 a0 = xp[lane];
  float4 a1 = xp[lane + 32];
  float s  = a0.x + a0.y + a0.z + a0.w + a1.x + a1.y + a1.z + a1.w;
  float s2 = a0.x*a0.x + a0.y*a0.y + a0.z*a0.z + a0.w*a0.w +
             a1.x*a1.x + a1.y*a1.y + a1.z*a1.z + a1.w*a1.w;
  for (int m = 16; m; m >>= 1) { s += __shfl_xor(s, m, 32); s2 += __shfl_xor(s2, m, 32); }
  float mu   = s * (1.0f / kC);
  float var  = s2 * (1.0f / kC) - mu * mu;
  float rstd = rsqrtf(var + 1e-5f);
  const float4* gp = (const float4*)g;
  const float4* bp = (const float4*)b;
  float4 g0 = gp[lane], g1 = gp[lane + 32];
  float4 b0 = bp[lane], b1 = bp[lane + 32];
  v4h h0, h1;
  h0[0] = (_Float16)((a0.x - mu) * rstd * g0.x + b0.x);
  h0[1] = (_Float16)((a0.y - mu) * rstd * g0.y + b0.y);
  h0[2] = (_Float16)((a0.z - mu) * rstd * g0.z + b0.z);
  h0[3] = (_Float16)((a0.w - mu) * rstd * g0.w + b0.w);
  h1[0] = (_Float16)((a1.x - mu) * rstd * g1.x + b1.x);
  h1[1] = (_Float16)((a1.y - mu) * rstd * g1.y + b1.y);
  h1[2] = (_Float16)((a1.z - mu) * rstd * g1.z + b1.z);
  h1[3] = (_Float16)((a1.w - mu) * rstd * g1.w + b1.w);
  v4h* op = (v4h*)(out + (size_t)row * kC);
  op[lane] = h0;
  op[lane + 32] = h1;
}

// ---------------------------------------------------------------------------
// WMMA GEMM with 2x4 register blocking: each wave computes 32 rows x 64 cols.
// Block = 8 waves (2 wave-rows x 4 wave-cols) -> 64 rows x 256 cols per block.
// grid (M/64, Nout/256).
// mode 0: f16 out (bias only)   mode 1: f32 out = val + residual
// mode 2: f32 out = gelu(val)
// ---------------------------------------------------------------------------
__global__ __launch_bounds__(256) void gemm16_kernel(
    const _Float16* __restrict__ A, const _Float16* __restrict__ packB,
    const float* __restrict__ bias, const int* __restrict__ gather,
    const float* __restrict__ residual, float* __restrict__ outF,
    _Float16* __restrict__ outH, int K, int Nout, int mode) {
  int lane = threadIdx.x & 31, wave = threadIdx.x >> 5;
  int NT = Nout >> 4;
  int half = lane >> 4, mr = lane & 15;
  int mt0 = blockIdx.x * 4 + (wave >> 2) * 2;    // this wave: mtiles mt0, mt0+1
  int nt0 = blockIdx.y * 16 + (wave & 3) * 4;    // this wave: ntiles nt0..nt0+3
  int row0 = mt0 * 16 + mr;
  int row1 = row0 + 16;
  int arow0 = gather ? gather[row0] : row0;
  int arow1 = gather ? gather[row1] : row1;
  const _Float16* ap0 = A + (size_t)arow0 * K + half * 16;
  const _Float16* ap1 = A + (size_t)arow1 * K + half * 16;
  v8f c[2][4] = {};
  int KT = K >> 5;
  for (int kt = 0; kt < KT; ++kt) {
    const _Float16* p0 = ap0 + kt * 32;
    const _Float16* p1 = ap1 + kt * 32;
    v16h a0 = mk16(*(const v8h*)p0, *(const v8h*)(p0 + 8));
    v16h a1 = mk16(*(const v8h*)p1, *(const v8h*)(p1 + 8));
    const _Float16* bbase = packB + ((((size_t)kt * NT + nt0) * 32 + lane) << 4);
#pragma unroll
    for (int j = 0; j < 4; ++j) {
      const _Float16* bp = bbase + (j << 9);     // 512 halves per ntile
      v16h b = mk16(*(const v8h*)bp, *(const v8h*)(bp + 8));
      c[0][j] = wmma_f16(a0, b, c[0][j]);
      c[1][j] = wmma_f16(a1, b, c[1][j]);
    }
  }
#pragma unroll
  for (int i = 0; i < 2; ++i) {
#pragma unroll
    for (int j = 0; j < 4; ++j) {
      int ncol = (nt0 + j) * 16 + mr;
      float bv = bias[ncol];
#pragma unroll
      for (int r = 0; r < 8; ++r) {
        int orow = (mt0 + i) * 16 + r + half * 8;
        size_t oi = (size_t)orow * Nout + ncol;
        float v = c[i][j][r] + bv;
        if (mode == 0) {
          outH[oi] = (_Float16)v;
        } else if (mode == 1) {
          outF[oi] = v + residual[oi];
        } else {
          outF[oi] = 0.5f * v * (1.0f + erff(v * 0.70710678118654752f));
        }
      }
    }
  }
}

// ---------------------------------------------------------------------------
// Patch attention: grid (P, H), 256 threads.
// qkv16 is [N x 768] f16 (patch gather via `order`). Output o16 [N x 256],
// patch-ordered (proj gathers with `inverse`). 1/sqrt(D) folded into the
// score write so q/k/v staging is a raw copy -> async-to-LDS eligible.
// ---------------------------------------------------------------------------
__global__ __launch_bounds__(256) void attn_kernel(const _Float16* __restrict__ qkv,
                                                   const int* __restrict__ order,
                                                   _Float16* __restrict__ o16) {
  extern __shared__ _Float16 smem[];
  _Float16* qs = smem;                         // 128*32
  _Float16* ks = qs + 4096;                    // 128*32
  _Float16* vs = ks + 4096;                    // 128*32
  float*    sc = (float*)(vs + 4096);          // 128*128 f32
  _Float16* at = (_Float16*)(sc + 16384);      // 128*128 f16

  int p = blockIdx.x, head = blockIdx.y;
  int t = threadIdx.x;
  const float scale = 0.17677669529663687f;    // 1/sqrt(32)

  // Stage q, k, v into LDS (raw f16 copies; q scaling deferred to scores).
  {
    int tok = t >> 1, part = t & 1;            // 16-half slice per thread
    int g = order[p * kPS + tok];
    const _Float16* src = qkv + (size_t)g * 768 + head * 32 + part * 16;
#pragma unroll
    for (int mtx = 0; mtx < 3; ++mtx) {        // q, k, v at +0, +256, +512
      const _Float16* s = src + mtx * 256;
      _Float16* d = qs + mtx * 4096 + tok * 32 + part * 16;
#ifdef HAVE_ASYNC_LDS
      async_copy16B(s, d);
      async_copy16B(s + 8, d + 8);
#else
      *(v8h*)d = *(const v8h*)s;
      *(v8h*)(d + 8) = *(const v8h*)(s + 8);
#endif
    }
  }
#ifdef HAVE_ASYNC_LDS
  async_wait0();
#endif
  __syncthreads();

  int lane = t & 31, wave = t >> 5;
  int half = lane >> 4, mr = lane & 15;

  // scores = (q @ k^T) * scale : wave = row tile, 8 col tiles, K=32 one WMMA.
  {
    v16h a = load_fragA(qs, 32, wave * 16, 0, lane);
#pragma unroll
    for (int ntj = 0; ntj < 8; ++ntj) {
      // B = k^T: column of B = row of k -> 16 contiguous halves per lane.
      const _Float16* bp = ks + (ntj * 16 + mr) * 32 + half * 16;
      v16h b = mk16(*(const v8h*)bp, *(const v8h*)(bp + 8));
      v8f c = {};
      c = wmma_f16(a, b, c);
#pragma unroll
      for (int r = 0; r < 8; ++r)
        sc[(wave * 16 + r + half * 8) * 128 + ntj * 16 + mr] = c[r] * scale;
    }
  }
  __syncthreads();

  // Row softmax (f32), result to f16 `at`.
  if (t < 128) {
    float mx = -1e30f;
    for (int j = 0; j < 128; ++j) mx = fmaxf(mx, sc[t * 128 + j]);
    float s = 0.f;
    for (int j = 0; j < 128; ++j) {
      float e = __expf(sc[t * 128 + j] - mx);
      sc[t * 128 + j] = e;
      s += e;
    }
    float inv = 1.f / s;
    for (int j = 0; j < 128; ++j) at[t * 128 + j] = (_Float16)(sc[t * 128 + j] * inv);
  }
  __syncthreads();

  // o = attn @ v : wave = row tile, 2 col tiles of 16 (D=32), 4 K-steps.
  for (int ntv = 0; ntv < 2; ++ntv) {
    v8f c = {};
#pragma unroll
    for (int kt = 0; kt < 4; ++kt) {
      v16h a = load_fragA(at, 128, wave * 16, kt * 32, lane);
      int d = ntv * 16 + mr;
      int kb = kt * 32 + half * 16;
      v16h b;
#pragma unroll
      for (int jj = 0; jj < 8; ++jj) {           // v rows strided by 32
        b[2 * jj]     = vs[(kb + 2 * jj) * 32 + d];
        b[2 * jj + 1] = vs[(kb + 2 * jj + 1) * 32 + d];
      }
      c = wmma_f16(a, b, c);
    }
#pragma unroll
    for (int r = 0; r < 8; ++r) {
      int i = wave * 16 + r + half * 8;
      o16[(size_t)(p * kPS + i) * kC + head * 32 + ntv * 16 + mr] = (_Float16)c[r];
    }
  }
}

// ---------------------------------------------------------------------------
// GRN: resp2[c] = sum_n m[n,c]^2 ; scale[c] = 1 + gamma[c]*rn[c]
// ---------------------------------------------------------------------------
__global__ void zero1024(float* r) { r[threadIdx.x] = 0.f; }

__global__ __launch_bounds__(256) void grn_partial(const float* __restrict__ m,
                                                   float* __restrict__ resp2) {
  int c = blockIdx.x * 256 + threadIdx.x;
  size_t rowbase = (size_t)blockIdx.y * 1024;
  float s = 0.f;
  for (int i = 0; i < 1024; ++i) {
    float v = m[(rowbase + i) * kHID + c];
    s += v * v;
  }
  atomicAdd(&resp2[c], s);
}

__global__ __launch_bounds__(1024) void grn_finalize(const float* __restrict__ resp2,
                                                     const float* __restrict__ gamma,
                                                     float* __restrict__ scale) {
  __shared__ float red[32];
  int c = threadIdx.x;
  float r = sqrtf(resp2[c]);
  float s = r;
  for (int m = 16; m; m >>= 1) s += __shfl_xor(s, m, 32);
  if ((c & 31) == 0) red[c >> 5] = s;
  __syncthreads();
  if (c < 32) {
    float v = red[c];
    for (int m = 16; m; m >>= 1) v += __shfl_xor(v, m, 32);
    if (c == 0) red[0] = v;
  }
  __syncthreads();
  float mean = red[0] * (1.0f / kHID);
  scale[c] = 1.0f + gamma[c] * (r / (mean + 1e-6f));
}

__global__ __launch_bounds__(256) void grn_apply(const float* __restrict__ m,
                                                 const float* __restrict__ scale,
                                                 const float* __restrict__ beta,
                                                 _Float16* __restrict__ m16) {
  size_t i = (size_t)blockIdx.x * 256 + threadIdx.x;
  int c = (int)(i & (kHID - 1));
  m16[i] = (_Float16)(m[i] * scale[c] + beta[c]);
}

// ---------------------------------------------------------------------------
extern "C" void kernel_launch(void* const* d_in, const int* in_sizes, int n_in,
                              void* d_out, int out_size, void* d_ws, size_t ws_size,
                              hipStream_t stream) {
  (void)in_sizes; (void)n_in; (void)out_size; (void)ws_size;
  const float* feat    = (const float*)d_in[0];
  const int*   order   = (const int*)d_in[2];
  const int*   inverse = (const int*)d_in[3];
  const float* w_qkv   = (const float*)d_in[4];
  const float* b_qkv   = (const float*)d_in[5];
  const float* w_proj  = (const float*)d_in[6];
  const float* b_proj  = (const float*)d_in[7];
  const float* ln1_g   = (const float*)d_in[8];
  const float* ln1_b   = (const float*)d_in[9];
  const float* ln2_g   = (const float*)d_in[10];
  const float* ln2_b   = (const float*)d_in[11];
  const float* w_fc1   = (const float*)d_in[12];
  const float* b_fc1   = (const float*)d_in[13];
  const float* w_fc2   = (const float*)d_in[14];
  const float* b_fc2   = (const float*)d_in[15];
  const float* grn_g   = (const float*)d_in[16];
  const float* grn_b   = (const float*)d_in[17];
  float* out = (float*)d_out;

  char* ws = (char*)d_ws;
  size_t off = 0;
  auto alloc = [&](size_t bytes) { char* p = ws + off; off += (bytes + 255) & ~(size_t)255; return p; };
  _Float16* pw_qkv  = (_Float16*)alloc((size_t)kC * 768 * 2);
  _Float16* pw_proj = (_Float16*)alloc((size_t)kC * kC * 2);
  _Float16* pw_fc1  = (_Float16*)alloc((size_t)kC * kHID * 2);
  _Float16* pw_fc2  = (_Float16*)alloc((size_t)kHID * kC * 2);
  _Float16* h16     = (_Float16*)alloc((size_t)kN * kC * 2);     // LN1 out, reused for LN2
  float*    x       = (float*)   alloc((size_t)kN * kC * 4);     // residual stream
  float*    m       = (float*)   alloc((size_t)kN * kHID * 4);   // fc1/gelu out
  _Float16* qkv16   = (_Float16*)alloc((size_t)kN * 768 * 2);
  _Float16* o16     = (_Float16*)alloc((size_t)kN * kC * 2);
  _Float16* m16     = (_Float16*)qkv16;                          // alias: qkv16+o16 dead by then
  float*    resp2   = (float*)   alloc(kHID * 4);
  float*    scale   = (float*)   alloc(kHID * 4);

  // 1. pack weights to B-fragment layout
  pack_b_kernel<<<(kC * 768 / 2) / 256, 256, 0, stream>>>(w_qkv, pw_qkv, kC, 768);
  pack_b_kernel<<<(kC * kC / 2) / 256, 256, 0, stream>>>(w_proj, pw_proj, kC, kC);
  pack_b_kernel<<<(kC * kHID / 2) / 256, 256, 0, stream>>>(w_fc1, pw_fc1, kC, kHID);
  pack_b_kernel<<<(kHID * kC / 2) / 256, 256, 0, stream>>>(w_fc2, pw_fc2, kHID, kC);

  // 2. h = LN1(feat) -> f16
  ln_kernel<<<kN / 8, 256, 0, stream>>>(feat, ln1_g, ln1_b, h16);

  // 3. qkv = h @ w_qkv + b_qkv -> f16
  gemm16_kernel<<<dim3(kN / 64, 768 / 256), 256, 0, stream>>>(
      h16, pw_qkv, b_qkv, nullptr, nullptr, nullptr, qkv16, kC, 768, 0);

  // 4. patch attention -> o16 (patch order)
  size_t smem = (size_t)3 * 4096 * 2 + 16384 * 4 + 16384 * 2;    // 120 KB
  attn_kernel<<<dim3(kP, kH), 256, smem, stream>>>(qkv16, order, o16);

  // 5. x = feat + o[inverse] @ w_proj + b_proj
  gemm16_kernel<<<dim3(kN / 64, kC / 256), 256, 0, stream>>>(
      o16, pw_proj, b_proj, inverse, feat, x, nullptr, kC, kC, 1);

  // 6. h2 = LN2(x) -> f16 (reuse h16)
  ln_kernel<<<kN / 8, 256, 0, stream>>>(x, ln2_g, ln2_b, h16);

  // 7. m = gelu(h2 @ w_fc1 + b_fc1) -> f32
  gemm16_kernel<<<dim3(kN / 64, kHID / 256), 256, 0, stream>>>(
      h16, pw_fc1, b_fc1, nullptr, nullptr, m, nullptr, kC, kHID, 2);

  // 8. GRN -> per-channel scale, apply + convert to f16
  zero1024<<<1, kHID, 0, stream>>>(resp2);
  grn_partial<<<dim3(kHID / 256, kN / 1024), 256, 0, stream>>>(m, resp2);
  grn_finalize<<<1, kHID, 0, stream>>>(resp2, grn_g, scale);
  grn_apply<<<((size_t)kN * kHID) / 256, 256, 0, stream>>>(m, scale, grn_b, m16);

  // 9. out = x + m16 @ w_fc2 + b_fc2
  gemm16_kernel<<<dim3(kN / 64, kC / 256), 256, 0, stream>>>(
      m16, pw_fc2, b_fc2, nullptr, x, out, nullptr, kHID, kC, 1);
}
